// ExperimentalGNN_4166118277636
// MI455X (gfx1250) — matrix operand order repaired and checked by reading, not compiled, not run on hardware
//
#include <hip/hip_runtime.h>
#include <hip/hip_bf16.h>
#include <math.h>

// ---------------------------------------------------------------------------
// ExperimentalGNN forward for MI455X (gfx1250, wave32).
// GEMMs use v_wmma_f32_16x16x32_bf16 with weights pre-packed to bf16 in the
// WMMA B-operand lane layout and staged into LDS via the Tensor Data Mover.
// Segment ops (GINE scatter, attention softmax, Set2Set) use float atomics.
// ---------------------------------------------------------------------------

#define NN 50000   // nodes      (multiple of 16)
#define EE 800000  // edges      (multiple of 16)
#define NG 128     // graphs

typedef __attribute__((ext_vector_type(16))) __bf16 v16bf;
typedef __attribute__((ext_vector_type(8)))  float  v8f;
typedef unsigned int u32x4 __attribute__((ext_vector_type(4)));
typedef int          i32x4 __attribute__((ext_vector_type(4)));
typedef int          i32x8 __attribute__((ext_vector_type(8)));

union BPack { v16bf v; uint4 q[2]; };

__device__ __forceinline__ float wsum(float v) {
#pragma unroll
  for (int m = 16; m > 0; m >>= 1) v += __shfl_xor(v, m, 32);
  return v;
}
__device__ __forceinline__ float sigm(float x) { return 1.f / (1.f + expf(-x)); }
__device__ __forceinline__ float silu_f(float x) { return x / (1.f + expf(-x)); }

__device__ __forceinline__ void atomicMaxF(float* addr, float val) {
  unsigned int* ua = (unsigned int*)addr;
  unsigned int old = *ua;
  while (__uint_as_float(old) < val) {
    unsigned int assumed = old;
    old = atomicCAS(ua, assumed, __float_as_uint(val));
    if (old == assumed) break;
  }
}

// Stage `dwords` DWORDs from global `src` into LDS `dst` (whole block must
// call; wave 0 issues one TDM descriptor, everyone joins the barrier).
__device__ __forceinline__ void tdm_stage(const unsigned* src, unsigned* dst, int dwords) {
#if __has_builtin(__builtin_amdgcn_tensor_load_to_lds)
  if (threadIdx.x < 32) {
    unsigned long long ga = (unsigned long long)src;
    unsigned ldsb = (unsigned)(unsigned long long)dst;  // low 32 bits = LDS offset
    u32x4 g0;
    g0[0] = 1u;                                   // count=1 (valid user D#)
    g0[1] = ldsb;                                 // lds_addr (bytes)
    g0[2] = (unsigned)(ga & 0xffffffffu);         // global_addr[31:0]
    g0[3] = (unsigned)((ga >> 32) & 0x01ffffffu)  // global_addr[56:32]
            | (2u << 30);                         // type=2 ("image")
    unsigned dw = (unsigned)dwords;
    i32x8 g1;
    g1[0] = (int)(2u << 16);                      // data_size=2 -> 4 bytes
    g1[1] = (int)((dw & 0xffffu) << 16);          // tensor_dim0[15:0]
    g1[2] = (int)(((dw >> 16) & 0xffffu) | (1u << 16)); // dim0 hi | tensor_dim1=1
    g1[3] = (int)((dw & 0xffffu) << 16);          // tile_dim0
    g1[4] = 1;                                    // tile_dim1=1 (tile_dim2=0)
    g1[5] = (int)dw;                              // tensor_dim0_stride[31:0]
    g1[6] = 0;
    g1[7] = 0;
    i32x4 gz = {0, 0, 0, 0};
#if defined(__clang_major__) && __clang_major__ >= 23
    i32x8 gz8 = {0, 0, 0, 0, 0, 0, 0, 0};
    __builtin_amdgcn_tensor_load_to_lds(g0, g1, gz, gz, gz8, 0);
#else
    __builtin_amdgcn_tensor_load_to_lds(g0, g1, gz, gz, 0);
#endif
    __builtin_amdgcn_s_wait_tensorcnt(0);
  }
  __syncthreads();
#else
  for (int i = threadIdx.x; i < dwords; i += blockDim.x) dst[i] = src[i];
  __syncthreads();
#endif
}

// ------------------------------ utility ------------------------------------
__global__ void fillf(float* p, float v, long n) {
  long i = (long)blockIdx.x * blockDim.x + threadIdx.x;
  if (i < n) p[i] = v;
}
__global__ void addv(float* y, const float* x, long n) {
  long i = (long)blockIdx.x * blockDim.x + threadIdx.x;
  if (i < n) y[i] += x[i];
}

// Pre-pack W[KR x 64] (f32, row-major) into bf16 WMMA B layout:
// pk[(t*KR + r)*8 + d] = { W[k][t*16+(r&15)] lo, W[k+1][t*16+(r&15)] hi },
// k = (r&~31) + ((d&4)<<2) + 8*((r>>4)&1) + 2*(d&3)
__global__ void pack_w(const float* __restrict__ W, unsigned* __restrict__ pk, int KR) {
  int i = blockIdx.x * blockDim.x + threadIdx.x;
  if (i >= KR * 32) return;            // 4 tiles * KR rows * 8 dwords
  int d = i & 7;
  int r = (i >> 3) % KR;
  int t = i / (KR * 8);
  int l = r & 31;
  int k = (r & ~31) + ((d & 4) << 2) + (l >> 4) * 8 + 2 * (d & 3);
  __bf16 b0 = (__bf16)W[k * 64 + t * 16 + (l & 15)];
  __bf16 b1 = (__bf16)W[(k + 1) * 64 + t * 16 + (l & 15)];
  unsigned short s0, s1;
  __builtin_memcpy(&s0, &b0, 2);
  __builtin_memcpy(&s1, &b1, 2);
  pk[i] = (unsigned)s0 | ((unsigned)s1 << 16);
}

// ------------------------------ encoders -----------------------------------
__global__ void node_encode(const float* __restrict__ x, const float* __restrict__ W,
                            const float* __restrict__ b, const float* __restrict__ g,
                            const float* __restrict__ bt, float* __restrict__ h) {
  long wv = ((long)blockIdx.x * blockDim.x + threadIdx.x) >> 5;
  int lane = threadIdx.x & 31;
  if (wv >= NN) return;
  const float* xr = x + wv * 11;
  float y0 = b[lane], y1 = b[lane + 32];
#pragma unroll
  for (int k = 0; k < 11; k++) {
    float xv = xr[k];
    y0 += xv * W[k * 64 + lane];
    y1 += xv * W[k * 64 + lane + 32];
  }
  float s = wsum(y0 + y1), s2 = wsum(y0 * y0 + y1 * y1);
  float mean = s * (1.f / 64.f);
  float var = s2 * (1.f / 64.f) - mean * mean;
  float r = rsqrtf(var + 1e-5f);
  float t0 = (y0 - mean) * r * g[lane] + bt[lane];
  float t1 = (y1 - mean) * r * g[lane + 32] + bt[lane + 32];
  h[wv * 64 + lane] = silu_f(t0);
  h[wv * 64 + lane + 32] = silu_f(t1);
}

__global__ void edge_encode(const float* __restrict__ ea, const float* __restrict__ W,
                            const float* __restrict__ b, const float* __restrict__ g,
                            const float* __restrict__ bt, float* __restrict__ eo) {
  long wv = ((long)blockIdx.x * blockDim.x + threadIdx.x) >> 5;
  int lane = threadIdx.x & 31;
  if (wv >= EE) return;
  const float* r3 = ea + wv * 3;
  float acc = b[lane] + r3[0] * W[lane] + r3[1] * W[32 + lane] + r3[2] * W[64 + lane];
  float s = wsum(acc), s2 = wsum(acc * acc);
  float mean = s * (1.f / 32.f);
  float var = s2 * (1.f / 32.f) - mean * mean;
  float t = (acc - mean) * rsqrtf(var + 1e-5f) * g[lane] + bt[lane];
  eo[wv * 32 + lane] = silu_f(t);
}

// ------------------------------ WMMA GEMMs ---------------------------------
// Y[E,64] = X[E,32] @ W(packed bf16) (+bias). One wave per 16-row tile; the
// 4KB packed weight block is TDM-staged into LDS once per workgroup.
__global__ void edge_gemm32_64(const float* __restrict__ X, const unsigned* __restrict__ pk,
                               const float* __restrict__ bias, float* __restrict__ Y,
                               int E, int has_bias) {
  __shared__ unsigned wlds[1024];  // 4 tiles * 32 rows * 8 dwords
  tdm_stage(pk, wlds, 1024);
  long wv = ((long)blockIdx.x * blockDim.x + threadIdx.x) >> 5;
  int lane = threadIdx.x & 31;
  long row0 = wv * 16;
  if (row0 > E - 16) row0 = E - 16;  // tail waves redo last tile (idempotent)
  const float* xr = X + (row0 + (lane & 15)) * 32;
  int kb = (lane >> 4) * 8;
  v16bf a;
#pragma unroll
  for (int q = 0; q < 8; q++) {
    a[q] = (__bf16)xr[kb + q];
    a[8 + q] = (__bf16)xr[16 + kb + q];
  }
  int mb = (lane >> 4) * 8;
  const uint4* pb = (const uint4*)wlds;
#pragma unroll
  for (int t = 0; t < 4; t++) {
    BPack B;
    B.q[0] = pb[(t * 32 + lane) * 2 + 0];
    B.q[1] = pb[(t * 32 + lane) * 2 + 1];
    v8f c = {};
    c = __builtin_amdgcn_wmma_f32_16x16x32_bf16(false, a, false, B.v, (short)0, c, false, false);
    int n = t * 16 + (lane & 15);
    float bv = has_bias ? bias[n] : 0.f;
#pragma unroll
    for (int rr = 0; rr < 8; rr++) Y[(row0 + mb + rr) * 64 + n] = c[rr] + bv;
  }
}

// Y[N,64] = X[N,64] @ W(packed bf16) (+bias). Two chained K=32 WMMAs per
// N-tile; 8KB packed weights TDM-staged to LDS per workgroup.
__global__ void node_gemm64(const float* __restrict__ X, const unsigned* __restrict__ pk,
                            const float* __restrict__ bias, float* __restrict__ Y,
                            int N, int has_bias) {
  __shared__ unsigned wlds[2048];  // 4 tiles * 64 rows * 8 dwords
  tdm_stage(pk, wlds, 2048);
  long wv = ((long)blockIdx.x * blockDim.x + threadIdx.x) >> 5;
  int lane = threadIdx.x & 31;
  long row0 = wv * 16;
  if (row0 > N - 16) row0 = N - 16;  // tail waves redo last tile (idempotent)
  const float* xr = X + (row0 + (lane & 15)) * 64;
  int kb = (lane >> 4) * 8;
  v16bf a0, a1;
#pragma unroll
  for (int q = 0; q < 8; q++) {
    a0[q] = (__bf16)xr[kb + q];
    a0[8 + q] = (__bf16)xr[16 + kb + q];
    a1[q] = (__bf16)xr[32 + kb + q];
    a1[8 + q] = (__bf16)xr[48 + kb + q];
  }
  int mb = (lane >> 4) * 8;
  const uint4* pb = (const uint4*)wlds;
#pragma unroll
  for (int t = 0; t < 4; t++) {
    BPack B0, B1;
    B0.q[0] = pb[(t * 64 + lane) * 2 + 0];
    B0.q[1] = pb[(t * 64 + lane) * 2 + 1];
    B1.q[0] = pb[(t * 64 + 32 + lane) * 2 + 0];
    B1.q[1] = pb[(t * 64 + 32 + lane) * 2 + 1];
    v8f c = {};
    c = __builtin_amdgcn_wmma_f32_16x16x32_bf16(false, a0, false, B0.v, (short)0, c, false, false);
    c = __builtin_amdgcn_wmma_f32_16x16x32_bf16(false, a1, false, B1.v, (short)0, c, false, false);
    int n = t * 16 + (lane & 15);
    float bv = has_bias ? bias[n] : 0.f;
#pragma unroll
    for (int rr = 0; rr < 8; rr++) Y[(row0 + mb + rr) * 64 + n] = c[rr] + bv;
  }
}

// ------------------------------ GINE ---------------------------------------
__global__ void gine_scatter(const float* __restrict__ h, const float* __restrict__ ee,
                             const int* __restrict__ eidx, float* __restrict__ agg, int E) {
  long i = (long)blockIdx.x * blockDim.x + threadIdx.x;
  if (i >= (long)E * 64) return;
  int e = (int)(i >> 6), c = (int)(i & 63);
  int s = eidx[e], d = eidx[E + e];
  float v = h[(long)s * 64 + c] + ee[(long)e * 64 + c];
  if (v > 0.f) atomicAdd(&agg[(long)d * 64 + c], v);
}

__global__ void ln_silu64(float* __restrict__ x, const float* __restrict__ g,
                          const float* __restrict__ b, int rows) {
  long wv = ((long)blockIdx.x * blockDim.x + threadIdx.x) >> 5;
  int lane = threadIdx.x & 31;
  if (wv >= rows) return;
  float* row = x + wv * 64;
  float y0 = row[lane], y1 = row[lane + 32];
  float s = wsum(y0 + y1), s2 = wsum(y0 * y0 + y1 * y1);
  float mean = s * (1.f / 64.f);
  float var = s2 * (1.f / 64.f) - mean * mean;
  float r = rsqrtf(var + 1e-5f);
  float t0 = (y0 - mean) * r * g[lane] + b[lane];
  float t1 = (y1 - mean) * r * g[lane + 32] + b[lane + 32];
  row[lane] = silu_f(t0);
  row[lane + 32] = silu_f(t1);
}

// ------------------------------ BatchNorm + residual -----------------------
__global__ void bn_stats(const float* __restrict__ x, float* __restrict__ sum,
                         float* __restrict__ sq, int N) {
  int tid = blockIdx.x * blockDim.x + threadIdx.x;
  int c = tid & 63;
  int row = tid >> 6;
  int stride = (gridDim.x * blockDim.x) >> 6;
  float s = 0.f, s2 = 0.f;
  for (int n = row; n < N; n += stride) {
    float v = x[(long)n * 64 + c];
    s += v;
    s2 += v * v;
  }
  atomicAdd(&sum[c], s);
  atomicAdd(&sq[c], s2);
}
__global__ void bn_apply_res(const float* __restrict__ hn, const float* __restrict__ sum,
                             const float* __restrict__ sq, const float* __restrict__ g,
                             const float* __restrict__ b, float* __restrict__ h, int N) {
  long i = (long)blockIdx.x * blockDim.x + threadIdx.x;
  if (i >= (long)N * 64) return;
  int c = (int)(i & 63);
  float m = sum[c] * (1.f / (float)N);
  float var = sq[c] * (1.f / (float)N) - m * m;
  float r = rsqrtf(var + 1e-5f);
  h[i] += (hn[i] - m) * r * g[c] + b[c];
}

// ------------------------------ TransformerConv ----------------------------
__global__ void attn_logits(const float* __restrict__ q, const float* __restrict__ k,
                            const float* __restrict__ ee, const int* __restrict__ eidx,
                            float* __restrict__ alpha, float* __restrict__ nmax, int E) {
  long i = (long)blockIdx.x * blockDim.x + threadIdx.x;
  if (i >= (long)E * 4) return;
  int e = (int)(i >> 2), hd = (int)(i & 3);
  int s = eidx[e], d = eidx[E + e];
  const float* qd = q + (long)d * 64 + hd * 16;
  const float* ks = k + (long)s * 64 + hd * 16;
  const float* eh = ee + (long)e * 64 + hd * 16;
  float acc = 0.f;
#pragma unroll
  for (int c = 0; c < 16; c++) acc += qd[c] * (ks[c] + eh[c]);
  acc *= 0.25f;  // 1/sqrt(16)
  alpha[i] = acc;
  atomicMaxF(&nmax[(long)d * 4 + hd], acc);
}
__global__ void attn_exp(float* __restrict__ alpha, const int* __restrict__ eidx,
                         const float* __restrict__ nmax, float* __restrict__ nsum, int E) {
  long i = (long)blockIdx.x * blockDim.x + threadIdx.x;
  if (i >= (long)E * 4) return;
  int e = (int)(i >> 2), hd = (int)(i & 3);
  int d = eidx[E + e];
  float ex = expf(alpha[i] - nmax[(long)d * 4 + hd]);
  alpha[i] = ex;
  atomicAdd(&nsum[(long)d * 4 + hd], ex);
}
__global__ void attn_agg(const float* __restrict__ v, const float* __restrict__ ee,
                         const float* __restrict__ alpha, const int* __restrict__ eidx,
                         const float* __restrict__ nsum, float* __restrict__ agg, int E) {
  long i = (long)blockIdx.x * blockDim.x + threadIdx.x;
  if (i >= (long)E * 64) return;
  int e = (int)(i >> 6), c = (int)(i & 63), hd = c >> 4;
  int s = eidx[e], d = eidx[E + e];
  float w = alpha[(long)e * 4 + hd] / (nsum[(long)d * 4 + hd] + 1e-16f);
  atomicAdd(&agg[(long)d * 64 + c], w * (v[(long)s * 64 + c] + ee[(long)e * 64 + c]));
}
__global__ void beta_combine(const float* __restrict__ agg, const float* __restrict__ xr,
                             const float* __restrict__ bW, float* __restrict__ hn, int N) {
  long wv = ((long)blockIdx.x * blockDim.x + threadIdx.x) >> 5;
  int lane = threadIdx.x & 31;
  if (wv >= N) return;
  float a0 = agg[wv * 64 + lane], a1 = agg[wv * 64 + lane + 32];
  float x0 = xr[wv * 64 + lane], x1 = xr[wv * 64 + lane + 32];
  float p = bW[lane] * a0 + bW[lane + 32] * a1 + bW[64 + lane] * x0 + bW[96 + lane] * x1 +
            bW[128 + lane] * (a0 - x0) + bW[160 + lane] * (a1 - x1);
  float beta = sigm(wsum(p));
  hn[wv * 64 + lane] = beta * x0 + (1.f - beta) * a0;
  hn[wv * 64 + lane + 32] = beta * x1 + (1.f - beta) * a1;
}

// ------------------------------ Set2Set ------------------------------------
__global__ void lstm_step(const float* __restrict__ qstar, const float* __restrict__ hs_in,
                          const float* __restrict__ cs_in, const float* __restrict__ Wih,
                          const float* __restrict__ Whh, const float* __restrict__ bl,
                          float* __restrict__ hs_out, float* __restrict__ cs_out) {
  int b = blockIdx.x;   // 0..127
  int j = threadIdx.x;  // 0..63
  float gi = bl[j], gf = bl[64 + j], gg = bl[128 + j], go = bl[192 + j];
  for (int k = 0; k < 128; k++) {
    float qv = qstar[b * 128 + k];
    const float* wr = Wih + k * 256;
    gi += qv * wr[j]; gf += qv * wr[64 + j]; gg += qv * wr[128 + j]; go += qv * wr[192 + j];
  }
  for (int k = 0; k < 64; k++) {
    float hv = hs_in[b * 64 + k];
    const float* wr = Whh + k * 256;
    gi += hv * wr[j]; gf += hv * wr[64 + j]; gg += hv * wr[128 + j]; go += hv * wr[192 + j];
  }
  float c = sigm(gf) * cs_in[b * 64 + j] + sigm(gi) * tanhf(gg);
  cs_out[b * 64 + j] = c;
  hs_out[b * 64 + j] = sigm(go) * tanhf(c);
}
__global__ void s2s_score(const float* __restrict__ h, const float* __restrict__ hs,
                          const int* __restrict__ batch, float* __restrict__ score,
                          float* __restrict__ amax, int N) {
  long wv = ((long)blockIdx.x * blockDim.x + threadIdx.x) >> 5;
  int lane = threadIdx.x & 31;
  if (wv >= N) return;
  int g = batch[wv];
  float p = h[wv * 64 + lane] * hs[(long)g * 64 + lane] +
            h[wv * 64 + lane + 32] * hs[(long)g * 64 + lane + 32];
  float s = wsum(p);
  if (lane == 0) {
    score[wv] = s;
    atomicMaxF(&amax[g], s);
  }
}
__global__ void s2s_exp(float* __restrict__ score, const int* __restrict__ batch,
                        const float* __restrict__ amax, float* __restrict__ asum, int N) {
  long i = (long)blockIdx.x * blockDim.x + threadIdx.x;
  if (i >= N) return;
  int g = batch[i];
  float ex = expf(score[i] - amax[g]);
  score[i] = ex;
  atomicAdd(&asum[g], ex);
}
__global__ void s2s_agg(const float* __restrict__ h, const float* __restrict__ score,
                        const int* __restrict__ batch, const float* __restrict__ asum,
                        float* __restrict__ rbuf, int N) {
  long i = (long)blockIdx.x * blockDim.x + threadIdx.x;
  if (i >= (long)N * 64) return;
  int n = (int)(i >> 6), c = (int)(i & 63);
  int g = batch[n];
  float w = score[n] / (asum[g] + 1e-16f);
  atomicAdd(&rbuf[(long)g * 64 + c], w * h[i]);
}
__global__ void qstar_build(const float* __restrict__ hs, const float* __restrict__ rbuf,
                            float* __restrict__ qstar) {
  int b = blockIdx.x, j = threadIdx.x;  // 128 x 128
  qstar[b * 128 + j] = (j < 64) ? hs[b * 64 + j] : rbuf[b * 64 + (j - 64)];
}

// ------------------------------ head ---------------------------------------
__global__ void glob1_ln(const float* __restrict__ nA, const float* __restrict__ nB,
                         const float* __restrict__ ss, const float* __restrict__ W1,
                         const float* __restrict__ b1, const float* __restrict__ g,
                         const float* __restrict__ bt, float* __restrict__ gvec) {
  long wv = ((long)blockIdx.x * blockDim.x + threadIdx.x) >> 5;
  int lane = threadIdx.x & 31;
  if (wv >= NG) return;
  float den = ss[wv] + 1e-10f;
  float f0 = nA[wv] / den, f1 = nB[wv] / den;
  float y0 = b1[lane] + f0 * W1[lane] + f1 * W1[64 + lane];
  float y1 = b1[lane + 32] + f0 * W1[lane + 32] + f1 * W1[96 + lane];
  float s = wsum(y0 + y1), s2 = wsum(y0 * y0 + y1 * y1);
  float mean = s * (1.f / 64.f);
  float var = s2 * (1.f / 64.f) - mean * mean;
  float r = rsqrtf(var + 1e-5f);
  float t0 = (y0 - mean) * r * g[lane] + bt[lane];
  float t1 = (y1 - mean) * r * g[lane + 32] + bt[lane + 32];
  gvec[wv * 64 + lane] = silu_f(t0);
  gvec[wv * 64 + lane + 32] = silu_f(t1);
}
__global__ void comb_build(const float* __restrict__ qstar, float* __restrict__ comb) {
  int b = blockIdx.x, j = threadIdx.x;  // 128 x 128
  comb[b * 192 + j] = qstar[b * 128 + j];
}
__global__ void glob2_comb(const float* __restrict__ gvec, const float* __restrict__ W2,
                           const float* __restrict__ b2, float* __restrict__ comb) {
  int b = blockIdx.x, oc = threadIdx.x;  // 128 x 64
  float acc = b2[oc];
  for (int k = 0; k < 64; k++) acc += gvec[b * 64 + k] * W2[k * 64 + oc];
  comb[b * 192 + 128 + oc] = acc;
}
__global__ void f1_gemm(const float* __restrict__ comb, const float* __restrict__ W,
                        const float* __restrict__ b, float* __restrict__ z1) {
  int bb = blockIdx.x, oc = threadIdx.x;  // 128 x 64
  float acc = b[oc];
  for (int k = 0; k < 192; k++) acc += comb[bb * 192 + k] * W[k * 64 + oc];
  z1[bb * 64 + oc] = acc;
}
__global__ void f2_gemm(const float* __restrict__ z1, const float* __restrict__ W,
                        const float* __restrict__ b, float* __restrict__ z2) {
  int bb = blockIdx.x, oc = threadIdx.x;  // 128 x 32
  float acc = b[oc];
  for (int k = 0; k < 64; k++) acc += z1[bb * 64 + k] * W[k * 32 + oc];
  z2[bb * 32 + oc] = acc;
}
__global__ void ln_silu32(float* __restrict__ x, const float* __restrict__ g,
                          const float* __restrict__ b, int rows) {
  long wv = ((long)blockIdx.x * blockDim.x + threadIdx.x) >> 5;
  int lane = threadIdx.x & 31;
  if (wv >= rows) return;
  float y = x[wv * 32 + lane];
  float s = wsum(y), s2 = wsum(y * y);
  float mean = s * (1.f / 32.f);
  float var = s2 * (1.f / 32.f) - mean * mean;
  float t = (y - mean) * rsqrtf(var + 1e-5f) * g[lane] + b[lane];
  x[wv * 32 + lane] = silu_f(t);
}
__global__ void f3_out(const float* __restrict__ z2, const float* __restrict__ W,
                       const float* __restrict__ b, float* __restrict__ out) {
  long wv = ((long)blockIdx.x * blockDim.x + threadIdx.x) >> 5;
  int lane = threadIdx.x & 31;
  if (wv >= NG) return;
  float s = wsum(z2[wv * 32 + lane] * W[lane]);
  if (lane == 0) out[wv] = s + b[0];
}

// ---------------------------------------------------------------------------
extern "C" void kernel_launch(void* const* d_in, const int* in_sizes, int n_in,
                              void* d_out, int out_size, void* d_ws, size_t ws_size,
                              hipStream_t stream) {
  (void)in_sizes; (void)n_in; (void)out_size; (void)ws_size;
  const float* x    = (const float*)d_in[0];
  const int*   eidx = (const int*)  d_in[1];   // [2,E]
  const float* ea   = (const float*)d_in[2];
  const int*   batch= (const int*)  d_in[3];
  const float* nA   = (const float*)d_in[4];
  const float* nB   = (const float*)d_in[5];
  const float* ss   = (const float*)d_in[6];
  // params pytree leaves (dict keys alphabetical, None biases dropped), base 7
#define PP(i) ((const float*)d_in[7 + (i)])

  float* w = (float*)d_ws;
  size_t o = 0;
  float* h    = w + o; o += (size_t)NN * 64;
  float* e    = w + o; o += (size_t)EE * 32;
  float* ee   = w + o; o += (size_t)EE * 64;
  float* agg  = w + o; o += (size_t)NN * 64;
  float* zb   = w + o; o += (size_t)NN * 64;
  float* hnew = w + o; o += (size_t)NN * 64;
  float* qb   = w + o; o += (size_t)NN * 64;
  float* kb   = w + o; o += (size_t)NN * 64;
  float* vb   = w + o; o += (size_t)NN * 64;
  float* sk   = w + o; o += (size_t)NN * 64;
  float* alph = w + o; o += (size_t)EE * 4;
  float* nmax = w + o; o += (size_t)NN * 4;
  float* nsum = w + o; o += (size_t)NN * 4;
  float* bnS  = w + o; o += 64;
  float* bnQ  = w + o; o += 64;
  float* hsA  = w + o; o += NG * 64;
  float* hsB  = w + o; o += NG * 64;
  float* csA  = w + o; o += NG * 64;
  float* csB  = w + o; o += NG * 64;
  float* qs   = w + o; o += NG * 128;
  float* rbuf = w + o; o += NG * 64;
  float* scr  = w + o; o += NN;
  float* amax = w + o; o += NG;
  float* asum = w + o; o += NG;
  float* gvec = w + o; o += NG * 64;
  float* comb = w + o; o += NG * 192;
  float* z1   = w + o; o += NG * 64;
  float* z2   = w + o; o += NG * 32;
  unsigned* pkN = (unsigned*)(w + o); o += 2048;  // packed 64x64 bf16 weights
  unsigned* pkE = (unsigned*)(w + o); o += 1024;  // packed 32x64 bf16 weights

  const int TB = 256;
  auto bt_ = [](long n) { return (unsigned)((n + 255) / 256); };   // thread-per-item
  auto bw_ = [](long n) { return (unsigned)((n + 7) / 8); };       // wave-per-item

  // pack + GEMM helpers (stream-ordered: pack scratch is reused per matrix)
  auto ngemm = [&](const float* X, const float* W64, const float* b, float* Y, int hb) {
    pack_w<<<bt_(64 * 32), TB, 0, stream>>>(W64, pkN, 64);
    node_gemm64<<<bw_(NN / 16), TB, 0, stream>>>(X, pkN, b, Y, NN, hb);
  };
  auto egemm = [&](const float* W32, const float* b, float* Y, int hb) {
    pack_w<<<bt_(32 * 32), TB, 0, stream>>>(W32, pkE, 32);
    edge_gemm32_64<<<bw_(EE / 16), TB, 0, stream>>>(e, pkE, b, Y, EE, hb);
  };

  // encoders (node_enc leaves at 111..114, edge_enc at 0..3)
  node_encode<<<bw_(NN), TB, 0, stream>>>(x, PP(113), PP(114), PP(112), PP(111), h);
  edge_encode<<<bw_(EE), TB, 0, stream>>>(ea, PP(2), PP(3), PP(1), PP(0), e);

  int base = 20;  // first layer leaf index
  for (int j = 0; j < 8; j++) {
    const float* bn_b;
    const float* bn_g;
    if ((j & 1) == 0) {  // GINEConv: bn_b,bn_g,lin_edge{W,b},ln_b,ln_g,mlp1{W,b},mlp2{W,b}
      bn_b = PP(base + 0); bn_g = PP(base + 1);
      egemm(PP(base + 2), PP(base + 3), ee, 1);
      fillf<<<bt_((long)NN * 64), TB, 0, stream>>>(agg, 0.f, (long)NN * 64);
      gine_scatter<<<bt_((long)EE * 64), TB, 0, stream>>>(h, ee, eidx, agg, EE);
      addv<<<bt_((long)NN * 64), TB, 0, stream>>>(agg, h, (long)NN * 64);
      ngemm(agg, PP(base + 6), PP(base + 7), zb, 1);
      ln_silu64<<<bw_(NN), TB, 0, stream>>>(zb, PP(base + 5), PP(base + 4), NN);
      ngemm(zb, PP(base + 8), PP(base + 9), hnew, 1);
      base += 10;
    } else {  // Transformer: betaW,bn_b,bn_g,eW,k{W,b},q{W,b},skip{W,b},v{W,b}
      bn_b = PP(base + 1); bn_g = PP(base + 2);
      ngemm(h, PP(base + 6), PP(base + 7), qb, 1);
      ngemm(h, PP(base + 4), PP(base + 5), kb, 1);
      ngemm(h, PP(base + 10), PP(base + 11), vb, 1);
      ngemm(h, PP(base + 8), PP(base + 9), sk, 1);
      egemm(PP(base + 3), nullptr, ee, 0);
      fillf<<<bt_((long)NN * 4), TB, 0, stream>>>(nmax, -3.0e38f, (long)NN * 4);
      fillf<<<bt_((long)NN * 4), TB, 0, stream>>>(nsum, 0.f, (long)NN * 4);
      fillf<<<bt_((long)NN * 64), TB, 0, stream>>>(agg, 0.f, (long)NN * 64);
      attn_logits<<<bt_((long)EE * 4), TB, 0, stream>>>(qb, kb, ee, eidx, alph, nmax, EE);
      attn_exp<<<bt_((long)EE * 4), TB, 0, stream>>>(alph, eidx, nmax, nsum, EE);
      attn_agg<<<bt_((long)EE * 64), TB, 0, stream>>>(vb, ee, alph, eidx, nsum, agg, EE);
      beta_combine<<<bw_(NN), TB, 0, stream>>>(agg, sk, PP(base + 0), hnew, NN);
      base += 12;
    }
    fillf<<<1, 64, 0, stream>>>(bnS, 0.f, 64);
    fillf<<<1, 64, 0, stream>>>(bnQ, 0.f, 64);
    bn_stats<<<256, TB, 0, stream>>>(hnew, bnS, bnQ, NN);
    bn_apply_res<<<bt_((long)NN * 64), TB, 0, stream>>>(hnew, bnS, bnQ, bn_g, bn_b, h, NN);
  }

  // Set2Set: lstm leaves Whh=108, Wih=109, b=110
  const float* Whh = PP(108);
  const float* Wih = PP(109);
  const float* bl = PP(110);
  fillf<<<bt_(NG * 64), TB, 0, stream>>>(hsA, 0.f, NG * 64);
  fillf<<<bt_(NG * 64), TB, 0, stream>>>(csA, 0.f, NG * 64);
  fillf<<<bt_(NG * 128), TB, 0, stream>>>(qs, 0.f, NG * 128);
  float* hin = hsA; float* hout = hsB; float* cin = csA; float* cout = csB;
  for (int it = 0; it < 4; it++) {
    lstm_step<<<NG, 64, 0, stream>>>(qs, hin, cin, Wih, Whh, bl, hout, cout);
    { float* t = hin; hin = hout; hout = t; t = cin; cin = cout; cout = t; }
    fillf<<<1, NG, 0, stream>>>(amax, -3.0e38f, NG);
    fillf<<<1, NG, 0, stream>>>(asum, 0.f, NG);
    fillf<<<bt_(NG * 64), TB, 0, stream>>>(rbuf, 0.f, NG * 64);
    s2s_score<<<bw_(NN), TB, 0, stream>>>(h, hin, batch, scr, amax, NN);
    s2s_exp<<<bt_(NN), TB, 0, stream>>>(scr, batch, amax, asum, NN);
    s2s_agg<<<bt_((long)NN * 64), TB, 0, stream>>>(h, scr, batch, asum, rbuf, NN);
    qstar_build<<<NG, 128, 0, stream>>>(hin, rbuf, qs);
  }

  // head: glob1=14,15 glob2=16,17 glob_b=18 glob_g=19 ; f1=4,5 f1_b=6 f1_g=7
  //       f2=8,9 f2_b=10 f2_g=11 ; f3=12,13
  glob1_ln<<<bw_(NG), TB, 0, stream>>>(nA, nB, ss, PP(14), PP(15), PP(19), PP(18), gvec);
  comb_build<<<NG, 128, 0, stream>>>(qs, comb);
  glob2_comb<<<NG, 64, 0, stream>>>(gvec, PP(16), PP(17), comb);
  f1_gemm<<<NG, 64, 0, stream>>>(comb, PP(4), PP(5), z1);
  ln_silu64<<<bw_(NG), TB, 0, stream>>>(z1, PP(7), PP(6), NG);
  f2_gemm<<<NG, 32, 0, stream>>>(z1, PP(8), PP(9), z2);
  ln_silu32<<<bw_(NG), TB, 0, stream>>>(z2, PP(11), PP(10), NG);
  f3_out<<<bw_(NG), TB, 0, stream>>>(z2, PP(12), PP(13), (float*)d_out);
#undef PP
}
